// GCN_attention_v2_58428735095548
// MI455X (gfx1250) — matrix operand order repaired and checked by reading, not compiled, not run on hardware
//
#include <hip/hip_runtime.h>

#define Nn     8192
#define NFEATc 512
#define NHIDc  128
#define NCLSc  16

typedef __attribute__((ext_vector_type(8)))  float  v8f;
typedef __attribute__((ext_vector_type(16))) __bf16 v16bf;
typedef __attribute__((ext_vector_type(8)))  __bf16 v8bf;
typedef __attribute__((ext_vector_type(4)))  __bf16 v4bf;

// ---------------------------------------------------------------------------
// 1) Attention logits + 2-way softmax: z1[i]=adj0[i,:]·w + b, z2 likewise.
//    One 256-thread block per row; float4 loads; LDS tree reduction.
// ---------------------------------------------------------------------------
__global__ void atten_kernel(const float* __restrict__ adj0,
                             const float* __restrict__ adj1,
                             const float* __restrict__ w,
                             const float* __restrict__ b,
                             float* __restrict__ nz0, float* __restrict__ nz1) {
  __shared__ float r0[256];
  __shared__ float r1[256];
  const int row = blockIdx.x;
  const float4* a0 = (const float4*)(adj0 + (size_t)row * Nn);
  const float4* a1 = (const float4*)(adj1 + (size_t)row * Nn);
  const float4* wv = (const float4*)w;
  float s0 = 0.f, s1 = 0.f;
  for (int i = threadIdx.x; i < Nn / 4; i += 256) {
    if (i + 256 < Nn / 4) {  // stream-ahead prefetch (global_prefetch_b8)
      __builtin_prefetch((const void*)(a0 + i + 256), 0, 0);
      __builtin_prefetch((const void*)(a1 + i + 256), 0, 0);
    }
    float4 ww = wv[i];
    float4 x0 = a0[i];
    float4 x1 = a1[i];
    s0 += x0.x * ww.x + x0.y * ww.y + x0.z * ww.z + x0.w * ww.w;
    s1 += x1.x * ww.x + x1.y * ww.y + x1.z * ww.z + x1.w * ww.w;
  }
  r0[threadIdx.x] = s0;
  r1[threadIdx.x] = s1;
  __syncthreads();
  for (int s = 128; s > 0; s >>= 1) {
    if ((int)threadIdx.x < s) {
      r0[threadIdx.x] += r0[threadIdx.x + s];
      r1[threadIdx.x] += r1[threadIdx.x + s];
    }
    __syncthreads();
  }
  if (threadIdx.x == 0) {
    float z0 = r0[0] + b[0];
    float z1 = r1[0] + b[0];
    float m  = fmaxf(z0, z1);
    float e0 = __expf(z0 - m);
    float e1 = __expf(z1 - m);
    float inv = 1.f / (e0 + e1);
    nz0[row] = e0 * inv;
    nz1[row] = e1 * inv;
  }
}

// ---------------------------------------------------------------------------
// 2) Blend: adjb[i,j] = bf16(nz0[j]*adj0[i,j] + nz1[j]*adj1[i,j])  (column wts)
// ---------------------------------------------------------------------------
__global__ void blend_kernel(const float* __restrict__ adj0,
                             const float* __restrict__ adj1,
                             const float* __restrict__ nz0,
                             const float* __restrict__ nz1,
                             __bf16* __restrict__ adjb) {
  const size_t total = (size_t)Nn * Nn / 4;
  const size_t stride = (size_t)gridDim.x * 256;
  for (size_t e = (size_t)blockIdx.x * 256 + threadIdx.x; e < total; e += stride) {
    const size_t nx = e + stride;
    if (nx < total) {
      __builtin_prefetch((const void*)(((const float4*)adj0) + nx), 0, 0);
      __builtin_prefetch((const void*)(((const float4*)adj1) + nx), 0, 0);
    }
    const size_t col4 = e & (Nn / 4 - 1);
    float4 x0 = ((const float4*)adj0)[e];
    float4 x1 = ((const float4*)adj1)[e];
    float4 w0 = ((const float4*)nz0)[col4];
    float4 w1 = ((const float4*)nz1)[col4];
    v4bf o;
    o[0] = (__bf16)(w0.x * x0.x + w1.x * x1.x);
    o[1] = (__bf16)(w0.y * x0.y + w1.y * x1.y);
    o[2] = (__bf16)(w0.z * x0.z + w1.z * x1.z);
    o[3] = (__bf16)(w0.w * x0.w + w1.w * x1.w);
    ((v4bf*)adjb)[e] = o;
  }
}

// ---------------------------------------------------------------------------
// 3) G0 = x @ W1, stored transposed in bf16: GT0[c][row].
// ---------------------------------------------------------------------------
__global__ void xw1_kernel(const float* __restrict__ x,
                           const float* __restrict__ W1,
                           __bf16* __restrict__ GT0) {
  __shared__ float xs[16 * NFEATc];  // 32 KB
  const int row0 = blockIdx.x * 16;
  const int c = threadIdx.x;  // 0..127
  for (int i = threadIdx.x; i < 16 * NFEATc; i += 128)
    xs[i] = x[(size_t)row0 * NFEATc + i];
  __syncthreads();
  float acc[16];
#pragma unroll
  for (int r = 0; r < 16; ++r) acc[r] = 0.f;
  for (int k = 0; k < NFEATc; ++k) {
    const float w = W1[(size_t)k * NHIDc + c];
#pragma unroll
    for (int r = 0; r < 16; ++r) acc[r] += xs[r * NFEATc + k] * w;
  }
#pragma unroll
  for (int r = 0; r < 16; ++r)
    GT0[(size_t)c * Nn + row0 + r] = (__bf16)acc[r];
}

// ---------------------------------------------------------------------------
// 4) G = h @ W (h: [Nn x 128] f32, W: [128 x NC]) stored transposed bf16.
// ---------------------------------------------------------------------------
__global__ void hwT_kernel(const float* __restrict__ h,
                           const float* __restrict__ W,
                           __bf16* __restrict__ GT, int NC) {
  __shared__ float hs[16 * NHIDc];  // 8 KB
  const int row0 = blockIdx.x * 16;
  const int c = threadIdx.x;  // 0..NC-1
  for (int i = threadIdx.x; i < 16 * NHIDc; i += blockDim.x)
    hs[i] = h[(size_t)row0 * NHIDc + i];
  __syncthreads();
  float acc[16];
#pragma unroll
  for (int r = 0; r < 16; ++r) acc[r] = 0.f;
  for (int k = 0; k < NHIDc; ++k) {
    const float w = W[(size_t)k * NC + c];
#pragma unroll
    for (int r = 0; r < 16; ++r) acc[r] += hs[r * NHIDc + k] * w;
  }
#pragma unroll
  for (int r = 0; r < 16; ++r)
    GT[(size_t)c * Nn + row0 + r] = (__bf16)acc[r];
}

// ---------------------------------------------------------------------------
// 5a) Big GEMM, Nout = 128, LDS-staged A:
//     Block = 8 waves covering 64 rows x 128 cols.  All 8 waves share the
//     same 64 A-rows, so the 64x128 bf16 A-chunk (16 KB) is staged in LDS
//     once per block per K-superstep (double buffered, 34 KB total), cutting
//     A L2 traffic 8x.  Row stride padded to 272 B (17 v8bf) so the 16
//     operand lanes land 4 banks apart.
//     Pipeline per superstep: issue global loads for chunk k+1 ->
//     16 WMMAs on chunk k from LDS (loads in flight) -> ds_store staged regs
//     -> barrier.
// ---------------------------------------------------------------------------
#define LDKV 17                 // row stride in v8bf units (136 halves)
#define ABUF (64 * LDKV)        // one buffer, in v8bf units

__global__ void __launch_bounds__(256)
adj_wmma_lds_kernel(const __bf16* __restrict__ A, const __bf16* __restrict__ BT,
                    const float* __restrict__ bias, float* __restrict__ C,
                    int relu) {
  __shared__ v8bf As[2 * ABUF];  // 34 KB
  const int tid = threadIdx.x;
  const int wv  = tid >> 5;
  const int lane = tid & 31;
  const int l16 = lane & 15;
  const int hlf = lane >> 4;
  const int m0 = blockIdx.x * 64;
  const int ncol = wv * 16 + l16;

  // per-thread staging coordinates (4 chunks of 16 B each = 64x128 halves)
  int srow[4], skq[4];
#pragma unroll
  for (int i = 0; i < 4; ++i) {
    const int c = tid + i * 256;
    srow[i] = c >> 4;
    skq[i]  = c & 15;
  }

  const __bf16* Bb = BT + (size_t)ncol * Nn + hlf * 16;

  v8f acc[4] = {};

  // prologue: stage chunk 0 into buffer 0
#pragma unroll
  for (int i = 0; i < 4; ++i)
    As[srow[i] * LDKV + skq[i]] =
        *(const v8bf*)(A + (size_t)(m0 + srow[i]) * Nn + skq[i] * 8);
  __syncthreads();

  const int NCHUNK = Nn / 128;  // 64
  for (int kc = 0; kc < NCHUNK; ++kc) {
    const int cur = kc & 1;

    // issue global loads for next chunk (kept in flight during compute)
    v8bf stage[4];
    if (kc + 1 < NCHUNK) {
      const size_t k0n = (size_t)(kc + 1) * 128;
#pragma unroll
      for (int i = 0; i < 4; ++i)
        stage[i] = *(const v8bf*)(A + (size_t)(m0 + srow[i]) * Nn + k0n + skq[i] * 8);
    }

    // compute 4 K-steps x 4 M-reps from LDS buffer `cur`
    const int k0 = kc * 128;
    const v8bf* Ab = &As[cur * ABUF];
#pragma unroll
    for (int kk = 0; kk < 128; kk += 32) {
      const v16bf bmat = *(const v16bf*)(Bb + k0 + kk);
#pragma unroll
      for (int r = 0; r < 4; ++r) {
        const int idx = (r * 16 + l16) * LDKV + (kk >> 3) + hlf * 2;
        const v8bf lo = Ab[idx];      // K = kk + hlf*16 + 0..7
        const v8bf hi = Ab[idx + 1];  // K = kk + hlf*16 + 8..15
        v16bf amat;
#pragma unroll
        for (int i = 0; i < 8; ++i) { amat[i] = lo[i]; amat[i + 8] = hi[i]; }
        acc[r] = __builtin_amdgcn_wmma_f32_16x16x32_bf16(
            false, amat, false, bmat, (short)0, acc[r], false, false);
      }
    }

    // commit next chunk to the other buffer (safe: it was fully consumed
    // last iteration and everyone passed the barrier since)
    if (kc + 1 < NCHUNK) {
      v8bf* dst = &As[(1 - cur) * ABUF];
#pragma unroll
      for (int i = 0; i < 4; ++i) dst[srow[i] * LDKV + skq[i]] = stage[i];
    }
    __syncthreads();
  }

  const float bn = bias[ncol];
#pragma unroll
  for (int r = 0; r < 4; ++r) {
#pragma unroll
    for (int i = 0; i < 8; ++i) {
      const int row = m0 + r * 16 + hlf * 8 + i;
      float v = acc[r][i] + bn;
      if (relu) v = fmaxf(v, 0.f);
      C[(size_t)row * NHIDc + ncol] = v;
    }
  }
}

// ---------------------------------------------------------------------------
// 5b) Big GEMM, generic (used for Nout = 16, where adjacency is read exactly
//     once so no cross-wave A redundancy exists).  Pure global b128 loads.
// ---------------------------------------------------------------------------
__global__ void __launch_bounds__(256)
adj_wmma_kernel(const __bf16* __restrict__ A, const __bf16* __restrict__ BT,
                const float* __restrict__ bias, float* __restrict__ C,
                int Nout, int relu) {
  const int wave = (blockIdx.x * 256 + threadIdx.x) >> 5;
  const int lane = threadIdx.x & 31;
  const int l16 = lane & 15;
  const int hlf = lane >> 4;
  const int ntiles = Nout >> 4;
  const int ntile = wave % ntiles;
  const int m0 = (wave / ntiles) * 64;
  const int ncol = ntile * 16 + l16;

  const __bf16* Abase = A + (size_t)(m0 + l16) * Nn + hlf * 16;
  const __bf16* Bbase = BT + (size_t)ncol * Nn + hlf * 16;

  v8f acc[4] = {};

  for (int k0 = 0; k0 < Nn; k0 += 32) {
    const v16bf bmat = *(const v16bf*)(Bbase + k0);
#pragma unroll
    for (int r = 0; r < 4; ++r) {
      const __bf16* pa = Abase + (size_t)r * 16 * Nn + k0;
      const v8bf lo = *(const v8bf*)pa;
      const v8bf hi = *(const v8bf*)(pa + 8);
      v16bf amat;
#pragma unroll
      for (int i = 0; i < 8; ++i) { amat[i] = lo[i]; amat[i + 8] = hi[i]; }
      acc[r] = __builtin_amdgcn_wmma_f32_16x16x32_bf16(
          false, amat, false, bmat, (short)0, acc[r], false, false);
    }
  }

  const float bn = bias[ncol];
#pragma unroll
  for (int r = 0; r < 4; ++r) {
#pragma unroll
    for (int i = 0; i < 8; ++i) {
      const int row = m0 + r * 16 + hlf * 8 + i;
      float v = acc[r][i] + bn;
      if (relu) v = fmaxf(v, 0.f);
      C[(size_t)row * Nout + ncol] = v;
    }
  }
}

// ---------------------------------------------------------------------------
// 6) Row softmax over 16 classes.
// ---------------------------------------------------------------------------
__global__ void softmax16_kernel(const float* __restrict__ z,
                                 float* __restrict__ out) {
  const int row = blockIdx.x * 256 + threadIdx.x;
  if (row >= Nn) return;
  const float* p = z + (size_t)row * NCLSc;
  float v[NCLSc];
  float m = p[0];
#pragma unroll
  for (int i = 0; i < NCLSc; ++i) {
    v[i] = p[i];
    m = fmaxf(m, v[i]);
  }
  float s = 0.f;
#pragma unroll
  for (int i = 0; i < NCLSc; ++i) {
    v[i] = __expf(v[i] - m);
    s += v[i];
  }
  const float inv = 1.f / s;
#pragma unroll
  for (int i = 0; i < NCLSc; ++i) out[(size_t)row * NCLSc + i] = v[i] * inv;
}

// ---------------------------------------------------------------------------
extern "C" void kernel_launch(void* const* d_in, const int* in_sizes, int n_in,
                              void* d_out, int out_size, void* d_ws,
                              size_t ws_size, hipStream_t stream) {
  const float* adj0 = (const float*)d_in[0];
  const float* adj1 = (const float*)d_in[1];
  const float* x    = (const float*)d_in[2];
  const float* aw   = (const float*)d_in[3];
  const float* ab   = (const float*)d_in[4];
  const float* W1   = (const float*)d_in[5];
  const float* b1   = (const float*)d_in[6];
  const float* Wm   = (const float*)d_in[7];
  const float* bm   = (const float*)d_in[8];
  const float* W2   = (const float*)d_in[9];
  const float* b2   = (const float*)d_in[10];
  float* out = (float*)d_out;

  char* ws = (char*)d_ws;
  size_t off = 0;
  __bf16* adjb = (__bf16*)(ws + off); off += (size_t)Nn * Nn * 2;       // 128 MB
  __bf16* GT0  = (__bf16*)(ws + off); off += (size_t)NHIDc * Nn * 2;    // 2 MB
  __bf16* GT1  = (__bf16*)(ws + off); off += (size_t)NHIDc * Nn * 2;    // 2 MB
  __bf16* GT2  = (__bf16*)(ws + off); off += (size_t)NCLSc * Nn * 2;    // 256 KB
  float*  h1   = (float*)(ws + off);  off += (size_t)Nn * NHIDc * 4;    // 4 MB
  float*  h2   = (float*)(ws + off);  off += (size_t)Nn * NHIDc * 4;    // 4 MB
  float*  zb   = (float*)(ws + off);  off += (size_t)Nn * NCLSc * 4;    // 512 KB
  float*  nz0  = (float*)(ws + off);  off += (size_t)Nn * 4;
  float*  nz1  = (float*)(ws + off);  off += (size_t)Nn * 4;

  // 1) attention weights
  atten_kernel<<<Nn, 256, 0, stream>>>(adj0, adj1, aw, ab, nz0, nz1);
  // 2) blended bf16 adjacency (L2-resident for the 3 GEMMs)
  blend_kernel<<<4096, 256, 0, stream>>>(adj0, adj1, nz0, nz1, adjb);
  // 3) G0^T = (x @ W1)^T
  xw1_kernel<<<Nn / 16, 128, 0, stream>>>(x, W1, GT0);
  // 4) h1 = relu(adj @ G0 + b1)   (LDS-staged A)
  adj_wmma_lds_kernel<<<Nn / 64, 256, 0, stream>>>(adjb, GT0, b1, h1, 1);
  // 5) G1^T = (h1 @ Wm)^T
  hwT_kernel<<<Nn / 16, NHIDc, 0, stream>>>(h1, Wm, GT1, NHIDc);
  // 6) h2 = relu(adj @ G1 + bm)   (LDS-staged A)
  adj_wmma_lds_kernel<<<Nn / 64, 256, 0, stream>>>(adjb, GT1, bm, h2, 1);
  // 7) G2^T = (h2 @ W2)^T
  hwT_kernel<<<Nn / 16, NCLSc, 0, stream>>>(h2, W2, GT2, NCLSc);
  // 8) z = adj @ G2 + b2          (adjacency read once -> pure global path)
  adj_wmma_kernel<<<(Nn / 64) * (NCLSc / 16) / 8, 256, 0, stream>>>(
      adjb, GT2, b2, zb, NCLSc, 0);
  // 9) softmax rows
  softmax16_kernel<<<Nn / 256, 256, 0, stream>>>(zb, out);
}
